// SUSAGEBin_64338610095087
// MI455X (gfx1250) — compile-verified, matrix operand order checked by reading, not run on hardware
//
#include <hip/hip_runtime.h>
#include <hip/hip_fp16.h>

typedef __attribute__((ext_vector_type(16))) _Float16 v16h;
typedef __attribute__((ext_vector_type(8)))  _Float16 v8h;
typedef __attribute__((ext_vector_type(8)))  float    v8f;

#define NNODES 100000
#define NEDGES 1600000
#define DFEAT  128
#define LROW   136   // padded LDS row stride in f16 (272 B) -> breaks 64-bank conflicts

// ---------------------------------------------------------------------------
// Degree: one float atomic per edge (computed once; degrees are layer-invariant)
// ---------------------------------------------------------------------------
__global__ __launch_bounds__(256) void deg_kernel(const int* __restrict__ dst,
                                                  float* __restrict__ deg, int E) {
    int e = blockIdx.x * blockDim.x + threadIdx.x;
    if (e < E) {
        __hip_atomic_fetch_add(&deg[dst[e]], 1.0f,
                               __ATOMIC_RELAXED, __HIP_MEMORY_SCOPE_AGENT);
    }
}

// ---------------------------------------------------------------------------
// Scatter-add aggregation: 32 threads per edge, float4 gather + 4 native
// global_atomic_add_f32 into agg[dst]. h/agg are L2-resident (51 MB << 192 MB).
// ---------------------------------------------------------------------------
__global__ __launch_bounds__(256) void scatter_kernel(const float* __restrict__ h,
                                                      const int* __restrict__ src,
                                                      const int* __restrict__ dst,
                                                      float* __restrict__ agg, int E) {
    int idx = blockIdx.x * blockDim.x + threadIdx.x;
    int e = idx >> 5;
    int c = (idx & 31) * 4;
    if (e < E) {
        int s = src[e];
        int d = dst[e];
        const float4 v = *(const float4*)(h + (size_t)s * DFEAT + c);
        float* p = agg + (size_t)d * DFEAT + c;
        __hip_atomic_fetch_add(p + 0, v.x, __ATOMIC_RELAXED, __HIP_MEMORY_SCOPE_AGENT);
        __hip_atomic_fetch_add(p + 1, v.y, __ATOMIC_RELAXED, __HIP_MEMORY_SCOPE_AGENT);
        __hip_atomic_fetch_add(p + 2, v.z, __ATOMIC_RELAXED, __HIP_MEMORY_SCOPE_AGENT);
        __hip_atomic_fetch_add(p + 3, v.w, __ATOMIC_RELAXED, __HIP_MEMORY_SCOPE_AGENT);
    }
}

// ---------------------------------------------------------------------------
// Pre-convert one layer's Wl,Wr (each do x di = 128x128 f32, row-major) into a
// single contiguous f16 buffer: wf16[0..16384) = Wl, wf16[16384..32768) = Wr.
// Done once per layer instead of per-wave-per-tile in the GEMM.
// ---------------------------------------------------------------------------
__global__ __launch_bounds__(256) void cvt_w_kernel(const float* __restrict__ Wl,
                                                    const float* __restrict__ Wr,
                                                    _Float16* __restrict__ wf16) {
    int i = blockIdx.x * blockDim.x + threadIdx.x;   // 0 .. 32767
    const int half = DFEAT * DFEAT;
    if (i < half)            wf16[i] = (_Float16)Wl[i];
    else if (i < 2 * half)   wf16[i] = (_Float16)Wr[i - half];
}

// ---------------------------------------------------------------------------
// Dense SAGE layer (di=do=128):
//   out = relu( (agg/deg) @ Wl^T + bl + h @ Wr^T )
//
// Per block (8 waves): stage both f16 weight matrices (64 KB) into LDS with
// gfx1250 async global->LDS copies (ASYNCcnt path), padded row stride to avoid
// bank conflicts, then each wave computes one 16-row tile: A operands
// (normalized agg + h, all K=128) live in registers; B tiles stream from LDS
// via ds_load_b128; 8 N-tiles x 4 K-steps x 2 matmuls = 64 WMMAs per wave.
//
// ISA VGPR layouts (CDNA5 ISA):
//   A 16x32 f16 : lanes 0-15 = rows M, hi lanes take K+8/K+24 groups
//   B 32x16 f16 : mirrored with N striped across lanes (B = W^T -> contiguous
//                 per-lane reads of W rows)
//   C/D 16x16 f32: VGPR v holds M = v + 8*(lane>>4), N = lane&15
// ---------------------------------------------------------------------------
__global__ __launch_bounds__(256) void sage_gemm_kernel(
    const float* __restrict__ h, const float* __restrict__ agg,
    const float* __restrict__ deg,
    const _Float16* __restrict__ wf16,      // [2][128][128] f16 (Wl, Wr)
    const float* __restrict__ bl,
    float* __restrict__ out, int n_nodes, int do_relu) {

    __shared__ __align__(16) _Float16 smem[2 * DFEAT * LROW];  // 68 KB

    // ---- async-stage 64 KB of f16 weights into LDS (row-padded) ----
    // 256 rows (Wl then Wr) x 16 chunks of 16 B; 4096 chunks / 256 thr = 16 iter
    {
        const uint32_t lbase = (uint32_t)(uintptr_t)smem;  // low 32 bits = LDS addr
#pragma unroll
        for (int j = 0; j < 16; ++j) {
            const int chunk = j * 256 + threadIdx.x;       // 0..4095
            const int row   = chunk >> 4;
            const int cb    = (chunk & 15) * 16;           // byte offset in row
            const uint32_t laddr = lbase + (uint32_t)(row * (LROW * 2) + cb);
            const char* g = (const char*)wf16 + chunk * 16;
            asm volatile("global_load_async_to_lds_b128 %0, %1, off"
                         :: "v"(laddr), "v"(g) : "memory");
        }
        asm volatile("s_wait_asynccnt 0" ::: "memory");
    }
    __syncthreads();

    const int lane = threadIdx.x & 31;
    const int wave = threadIdx.x >> 5;
    const int tile = blockIdx.x * 8 + wave;
    const int row0 = tile * 16;
    if (row0 >= n_nodes) return;

    const int m  = row0 + (lane & 15);
    const int hi = lane >> 4;                       // K half select
    const float scale = 1.0f / fmaxf(deg[m], 1.0f); // mean-aggregation, folded

    const float* __restrict__ hrow = h   + (size_t)m * DFEAT;
    const float* __restrict__ arow = agg + (size_t)m * DFEAT;

    // Preload + convert A operands for all 4 K-tiles (K = 128)
    v16h aggA[4], hA[4];
#pragma unroll
    for (int kt = 0; kt < 4; ++kt) {
        const int kb = kt * 32 + hi * 8;
#pragma unroll
        for (int i = 0; i < 8; ++i) {
            aggA[kt][i]     = (_Float16)(arow[kb + i]      * scale);
            aggA[kt][8 + i] = (_Float16)(arow[kb + 16 + i] * scale);
            hA[kt][i]       = (_Float16) hrow[kb + i];
            hA[kt][8 + i]   = (_Float16) hrow[kb + 16 + i];
        }
    }

    const _Float16* __restrict__ lwl = smem;                 // Wl rows
    const _Float16* __restrict__ lwr = smem + DFEAT * LROW;  // Wr rows

#pragma unroll 1
    for (int nt = 0; nt < 8; ++nt) {
        const int n = nt * 16 + (lane & 15);
        const float bias = bl[n];
        v8f c;
#pragma unroll
        for (int v = 0; v < 8; ++v) c[v] = bias;

        const _Float16* wl = lwl + n * LROW;
        const _Float16* wr = lwr + n * LROW;
#pragma unroll
        for (int kt = 0; kt < 4; ++kt) {
            const int kb = kt * 32 + hi * 8;
            const v8h bl_lo = *(const v8h*)(wl + kb);
            const v8h bl_hi = *(const v8h*)(wl + kb + 16);
            const v8h br_lo = *(const v8h*)(wr + kb);
            const v8h br_hi = *(const v8h*)(wr + kb + 16);
            v16h Blt, Brt;
#pragma unroll
            for (int i = 0; i < 8; ++i) {
                Blt[i] = bl_lo[i]; Blt[8 + i] = bl_hi[i];
                Brt[i] = br_lo[i]; Brt[8 + i] = br_hi[i];
            }
            c = __builtin_amdgcn_wmma_f32_16x16x32_f16(
                    false, aggA[kt], false, Blt, (short)0, c, false, false);
            c = __builtin_amdgcn_wmma_f32_16x16x32_f16(
                    false, hA[kt],   false, Brt, (short)0, c, false, false);
        }
#pragma unroll
        for (int v = 0; v < 8; ++v) {
            float val = c[v];
            if (do_relu) val = fmaxf(val, 0.0f);
            out[(size_t)(row0 + v + 8 * hi) * DFEAT + n] = val;
        }
    }
}

// ---------------------------------------------------------------------------
// Final layer (128 -> 1) is a GEMV: WMMA would waste 15/16 of the tile, so use
// one wave per node: float4 loads, fused dot of both terms, shfl_xor reduce.
// Writes d_out[0..N) = h and d_out[N..2N) = sigmoid(h).
// ---------------------------------------------------------------------------
__global__ __launch_bounds__(256) void sage_out_kernel(
    const float* __restrict__ h, const float* __restrict__ agg,
    const float* __restrict__ deg,
    const float* __restrict__ Wl, const float* __restrict__ bl,
    const float* __restrict__ Wr,
    float* __restrict__ out, int n_nodes) {

    const int lane = threadIdx.x & 31;
    const int node = blockIdx.x * 8 + (threadIdx.x >> 5);
    if (node >= n_nodes) return;

    const float scale = 1.0f / fmaxf(deg[node], 1.0f);
    const float4 a  = *(const float4*)(agg + (size_t)node * DFEAT + lane * 4);
    const float4 x  = *(const float4*)(h   + (size_t)node * DFEAT + lane * 4);
    const float4 wl = *(const float4*)(Wl + lane * 4);
    const float4 wr = *(const float4*)(Wr + lane * 4);

    float acc = scale * (a.x * wl.x + a.y * wl.y + a.z * wl.z + a.w * wl.w)
              + x.x * wr.x + x.y * wr.y + x.z * wr.z + x.w * wr.w;
#pragma unroll
    for (int off = 16; off > 0; off >>= 1)
        acc += __shfl_xor(acc, off, 32);

    if (lane == 0) {
        const float val = acc + bl[0];
        out[node]           = val;
        out[n_nodes + node] = 1.0f / (1.0f + __expf(-val));
    }
}

// ---------------------------------------------------------------------------
extern "C" void kernel_launch(void* const* d_in, const int* in_sizes, int n_in,
                              void* d_out, int out_size, void* d_ws, size_t ws_size,
                              hipStream_t stream) {
    const float* x   = (const float*)d_in[0];
    const int*   ei  = (const int*)d_in[1];
    const int*   src = ei;            // edge_index row 0
    const int*   dst = ei + NEDGES;   // edge_index row 1
    const float* Wl0 = (const float*)d_in[2];
    const float* bl0 = (const float*)d_in[3];
    const float* Wr0 = (const float*)d_in[4];
    const float* Wl1 = (const float*)d_in[5];
    const float* bl1 = (const float*)d_in[6];
    const float* Wr1 = (const float*)d_in[7];
    const float* Wl2 = (const float*)d_in[8];
    const float* bl2 = (const float*)d_in[9];
    const float* Wr2 = (const float*)d_in[10];
    float* out = (float*)d_out;

    const size_t featBytes = (size_t)NNODES * DFEAT * sizeof(float); // 51.2 MB
    const size_t degBytes  = (((size_t)NNODES * sizeof(float)) + 255) & ~(size_t)255;
    const size_t wBytes    = 2 * DFEAT * DFEAT * sizeof(_Float16);   // 64 KB

    char* p    = (char*)d_ws;
    float* deg = (float*)p;                 p += degBytes;
    float* agg = (float*)p;                 p += featBytes;
    float* hA  = (float*)p;                 p += featBytes;
    float* hB  = (float*)p;                 p += featBytes;
    _Float16* wf16 = (_Float16*)p;          p += wBytes;

    const int degBlocks     = (NEDGES + 255) / 256;
    const int scatterBlocks = (NEDGES * 32) / 256;          // exact: 200000
    const int gemmBlocks    = ((NNODES / 16) + 7) / 8;      // 16-row tiles, 8 waves/blk
    const int outBlocks     = (NNODES + 7) / 8;
    const int cvtBlocks     = (2 * DFEAT * DFEAT) / 256;    // 128

    // degrees (once)
    hipMemsetAsync(deg, 0, (size_t)NNODES * sizeof(float), stream);
    deg_kernel<<<degBlocks, 256, 0, stream>>>(dst, deg, NEDGES);

    // layer 0: x -> hA
    hipMemsetAsync(agg, 0, featBytes, stream);
    scatter_kernel<<<scatterBlocks, 256, 0, stream>>>(x, src, dst, agg, NEDGES);
    cvt_w_kernel<<<cvtBlocks, 256, 0, stream>>>(Wl0, Wr0, wf16);
    sage_gemm_kernel<<<gemmBlocks, 256, 0, stream>>>(x, agg, deg, wf16, bl0,
                                                     hA, NNODES, 1);

    // layer 1: hA -> hB
    hipMemsetAsync(agg, 0, featBytes, stream);
    scatter_kernel<<<scatterBlocks, 256, 0, stream>>>(hA, src, dst, agg, NEDGES);
    cvt_w_kernel<<<cvtBlocks, 256, 0, stream>>>(Wl1, Wr1, wf16);
    sage_gemm_kernel<<<gemmBlocks, 256, 0, stream>>>(hA, agg, deg, wf16, bl1,
                                                     hB, NNODES, 1);

    // layer 2 (128 -> 1): hB -> (h, sigmoid(h)) in d_out
    hipMemsetAsync(agg, 0, featBytes, stream);
    scatter_kernel<<<scatterBlocks, 256, 0, stream>>>(hB, src, dst, agg, NEDGES);
    sage_out_kernel<<<outBlocks, 256, 0, stream>>>(hB, agg, deg, Wl2, bl2, Wr2,
                                                   out, NNODES);
}
